// LLMSelector_91070486545013
// MI455X (gfx1250) — matrix operand order repaired and checked by reading, not compiled, not run on hardware
//
#include <hip/hip_runtime.h>
#include <hip/hip_bf16.h>
#include <math.h>

// ---------------------------------------------------------------------------
// LLMSelector for MI455X (gfx1250): WMMA f16 GEMMs with async-to-LDS double
// buffering for the big GEMM; bandwidth-streaming kernels for GEMV / scatter.
// ---------------------------------------------------------------------------

typedef __attribute__((ext_vector_type(16))) _Float16 v16h;
typedef __attribute__((ext_vector_type(8)))  float    v8f;

#define NR    32768   // N_ROLES
#define DIM   512     // IN_DIM
#define HID   512     // HIDDEN
#define NE    524288  // N_EDGES
#define NL    64      // N_LLMS

// ---------------------------------------------------------------------------
// helpers
// ---------------------------------------------------------------------------
__device__ __forceinline__ float wave_reduce_add(float v) {
#pragma unroll
    for (int m = 1; m < 32; m <<= 1) v += __shfl_xor(v, m, 32);
    return v;
}

__device__ __forceinline__ float hash_u01(unsigned x) {
    x ^= x >> 17; x *= 0xed5ad4bbu;
    x ^= x >> 11; x *= 0xac4c1b51u;
    x ^= x >> 15; x *= 0x31848babu;
    x ^= x >> 14;
    return (float)(x >> 8) * (1.0f / 16777216.0f) + 1.0e-7f; // (0,1)
}

// gfx1250 async global->LDS copy (ASYNCcnt-tracked, no VGPR data round-trip)
__device__ __forceinline__ void async_copy_b128(unsigned lds_addr, const float* gptr) {
    asm volatile("global_load_async_to_lds_b128 %0, %1, off"
                 :: "v"(lds_addr), "v"(gptr) : "memory");
}
__device__ __forceinline__ void async_wait0() {
    asm volatile("s_wait_asynccnt 0x0" ::: "memory");
}

// ---------------------------------------------------------------------------
// 0) zero agg, deg, log-prob accumulator (must run every launch: atomics)
// ---------------------------------------------------------------------------
__global__ void k_init(float* agg, float* deg, float* out_logp) {
    const size_t tid    = (size_t)blockIdx.x * blockDim.x + threadIdx.x;
    const size_t stride = (size_t)gridDim.x * blockDim.x;
    for (size_t i = tid; i < (size_t)NR * HID; i += stride) agg[i] = 0.0f;
    for (size_t i = tid; i < NR; i += stride)               deg[i] = 0.0f;
    if (tid == 0) out_logp[0] = 0.0f;
}

// ---------------------------------------------------------------------------
// 1) llm_raw = llm_emb @ W_llm^T + b   (M=64, N=512, K=512) -- WMMA f16
//    grid(nt=32, mt=4), one wave per block (tiny problem; direct loads)
// ---------------------------------------------------------------------------
__global__ void k_llm_gemm(const float* __restrict__ emb,
                           const float* __restrict__ W,
                           const float* __restrict__ b,
                           float* __restrict__ out) {
    const int lane = threadIdx.x & 31;
    const int lo = lane & 15, hi = lane >> 4;
    const int nt = blockIdx.x, mt = blockIdx.y;

    const float* arow = emb + (size_t)(mt * 16 + lo) * DIM;
    const float* brow = W   + (size_t)(nt * 16 + lo) * DIM;

    v8f c = {};
    for (int kb = 0; kb < DIM; kb += 32) {
        v16h a, bf;
#pragma unroll
        for (int i = 0; i < 16; ++i) {
            const int v = i >> 1, p = i & 1;
            const int ka = ((v < 4) ? (2 * v) : (16 + 2 * (v - 4))) + hi * 8 + p;
            a[i] = (_Float16)arow[kb + ka];
            const int kk = ((v < 4) ? (2 * v) : (16 + 2 * (v - 4))) + hi * 8 + p;
            bf[i] = (_Float16)brow[kb + kk];
        }
        c = __builtin_amdgcn_wmma_f32_16x16x32_f16(false, a, false, bf,
                                                   (short)0, c, false, false);
    }
#pragma unroll
    for (int r = 0; r < 8; ++r) {
        const int m = mt * 16 + r + hi * 8;
        const int n = nt * 16 + lo;
        out[(size_t)m * HID + n] = c[r] + b[n];
    }
}

// ---------------------------------------------------------------------------
// 2) in-place row L2 normalize (width 512), one wave per row
// ---------------------------------------------------------------------------
__global__ void k_rownorm(float* __restrict__ buf, int nrows) {
    const int wid  = (blockIdx.x * blockDim.x + threadIdx.x) >> 5;
    const int lane = threadIdx.x & 31;
    if (wid >= nrows) return;
    float* p = buf + (size_t)wid * HID + lane * 16;
    float4 v0 = *(const float4*)(p + 0);
    float4 v1 = *(const float4*)(p + 4);
    float4 v2 = *(const float4*)(p + 8);
    float4 v3 = *(const float4*)(p + 12);
    float ss = v0.x*v0.x + v0.y*v0.y + v0.z*v0.z + v0.w*v0.w
             + v1.x*v1.x + v1.y*v1.y + v1.z*v1.z + v1.w*v1.w
             + v2.x*v2.x + v2.y*v2.y + v2.z*v2.z + v2.w*v2.w
             + v3.x*v3.x + v3.y*v3.y + v3.z*v3.z + v3.w*v3.w;
    ss = wave_reduce_add(ss);
    const float inv = 1.0f / fmaxf(sqrtf(ss), 1.0e-12f);
    v0.x*=inv; v0.y*=inv; v0.z*=inv; v0.w*=inv;
    v1.x*=inv; v1.y*=inv; v1.z*=inv; v1.w*=inv;
    v2.x*=inv; v2.y*=inv; v2.z*=inv; v2.w*=inv;
    v3.x*=inv; v3.y*=inv; v3.z*=inv; v3.w*=inv;
    *(float4*)(p + 0)  = v0; *(float4*)(p + 4)  = v1;
    *(float4*)(p + 8)  = v2; *(float4*)(p + 12) = v3;
}

// ---------------------------------------------------------------------------
// 3) edge weights: ew = relu(edge_emb @ W_ew^T + b); degree accumulation;
//    edge-index passthrough.  One wave per edge (streams the 1 GB input).
// ---------------------------------------------------------------------------
__global__ void k_edge(const float* __restrict__ eemb,
                       const float* __restrict__ Wew, const float* __restrict__ bew,
                       const int* __restrict__ ei,
                       float* __restrict__ out_ew, float* __restrict__ out_ei,
                       float* __restrict__ deg) {
    const int e    = (blockIdx.x * blockDim.x + threadIdx.x) >> 5;
    const int lane = threadIdx.x & 31;
    if (e >= NE) return;
    const float* a = eemb + (size_t)e * DIM + lane * 16;
    const float* w = Wew + lane * 16;
    float d = 0.0f;
#pragma unroll
    for (int j = 0; j < 4; ++j) {
        const float4 av = *(const float4*)(a + 4 * j);
        const float4 wv = *(const float4*)(w + 4 * j);
        d += av.x * wv.x + av.y * wv.y + av.z * wv.z + av.w * wv.w;
    }
    d = wave_reduce_add(d);
    if (lane == 0) {
        const float ew = fmaxf(d + bew[0], 0.0f);
        out_ew[e] = ew;
        const int r = ei[e], c = ei[NE + e];
        out_ei[e]      = (float)r;
        out_ei[NE + e] = (float)c;
        atomicAdd(&deg[c], ew);
    }
}

// ---------------------------------------------------------------------------
// 4) dinv = rsqrt(deg + 1)   (self-loop weight 1 included here)
// ---------------------------------------------------------------------------
__global__ void k_dinv(const float* __restrict__ deg, float* __restrict__ dinv) {
    const int n = blockIdx.x * blockDim.x + threadIdx.x;
    if (n >= NR) return;
    const float d = deg[n] + 1.0f;
    dinv[n] = (d > 0.0f) ? rsqrtf(fmaxf(d, 1.0e-30f)) : 0.0f;
}

// ---------------------------------------------------------------------------
// 5) x = [role | query] @ W_gcn^T   (M=32768, N=512, K=1024) -- WMMA f16
//    Block tile 128x64, K-step 32, async global->LDS double buffering.
//    8 waves/block; wave w owns rows [w*16, w*16+16) x all 64 cols (4 WMMA).
//    LDS: A 128x32 f32 (16KB) + B 64x32 f32 (8KB), x2 buffers = 48KB.
// ---------------------------------------------------------------------------
#define BM 128
#define BN 64
#define BK 32
#define KSTEPS ((2 * DIM) / BK)

__global__ void __launch_bounds__(256)
k_gemm_rq(const float* __restrict__ role,
          const float* __restrict__ query,
          const float* __restrict__ Wg,
          float* __restrict__ x) {
    __shared__ __align__(16) float sA[2][BM * BK];
    __shared__ __align__(16) float sB[2][BN * BK];

    const int tid  = threadIdx.x;
    const int lane = tid & 31;
    const int wv   = tid >> 5;          // 0..7
    const int lo = lane & 15, hi = lane >> 4;
    const int n0 = blockIdx.x * BN;
    const int m0 = blockIdx.y * BM;

    // per-thread tile-fill coordinates
    const int arow_t = tid >> 1;              // 0..127
    const int akk0   = (tid & 1) * 16;        // 0 or 16  (16 floats = 4x b128)
    const int brow_t = tid >> 2;              // 0..63
    const int bkk0   = (tid & 3) * 8;         // 0,8,16,24 (8 floats = 2x b128)

    // issue async fills for K-step `s` into buffer `buf`
    auto issue = [&](int buf, int s) {
        const int kb = s * BK;
#pragma unroll
        for (int j = 0; j < 4; ++j) {
            const int k = kb + akk0 + 4 * j;
            const float* g = (k < DIM)
                ? (role + (size_t)(m0 + arow_t) * DIM + k)
                : (query + (k - DIM));
            async_copy_b128((unsigned)(uintptr_t)&sA[buf][arow_t * BK + akk0 + 4 * j], g);
        }
#pragma unroll
        for (int j = 0; j < 2; ++j) {
            const int k = kb + bkk0 + 4 * j;
            const float* g = Wg + (size_t)(n0 + brow_t) * (2 * DIM) + k;
            async_copy_b128((unsigned)(uintptr_t)&sB[buf][brow_t * BK + bkk0 + 4 * j], g);
        }
    };

    v8f c0 = {}, c1 = {}, c2 = {}, c3 = {};

    issue(0, 0);
    for (int s = 0; s < KSTEPS; ++s) {
        const int buf = s & 1;
        async_wait0();          // my async fills for `buf` have landed
        __syncthreads();        // everyone's have landed
        if (s + 1 < KSTEPS) issue(buf ^ 1, s + 1);

        // A fragment: row = wv*16+lo, K pairs per ISA 16x32 f16 layout
        const float* Ar = &sA[buf][(wv * 16 + lo) * BK];
        v16h a;
#pragma unroll
        for (int v = 0; v < 8; ++v) {
            const int ka = ((v < 4) ? (2 * v) : (16 + 2 * (v - 4))) + hi * 8;
            const float2 f = *(const float2*)(Ar + ka);
            a[2 * v]     = (_Float16)f.x;
            a[2 * v + 1] = (_Float16)f.y;
        }
        // B fragments for the 4 N-tiles
        v16h bf[4];
#pragma unroll
        for (int nt = 0; nt < 4; ++nt) {
            const float* Br = &sB[buf][(nt * 16 + lo) * BK];
#pragma unroll
            for (int v = 0; v < 8; ++v) {
                const int kk = ((v < 4) ? (2 * v) : (16 + 2 * (v - 4))) + hi * 8;
                const float2 f = *(const float2*)(Br + kk);
                bf[nt][2 * v]     = (_Float16)f.x;
                bf[nt][2 * v + 1] = (_Float16)f.y;
            }
        }
        c0 = __builtin_amdgcn_wmma_f32_16x16x32_f16(false, a, false, bf[0], (short)0, c0, false, false);
        c1 = __builtin_amdgcn_wmma_f32_16x16x32_f16(false, a, false, bf[1], (short)0, c1, false, false);
        c2 = __builtin_amdgcn_wmma_f32_16x16x32_f16(false, a, false, bf[2], (short)0, c2, false, false);
        c3 = __builtin_amdgcn_wmma_f32_16x16x32_f16(false, a, false, bf[3], (short)0, c3, false, false);
        __syncthreads();        // done reading `buf` before it is refilled
    }

#pragma unroll
    for (int r = 0; r < 8; ++r) {
        const int m = m0 + wv * 16 + r + hi * 8;
        float* xp = x + (size_t)m * HID + n0 + lo;
        xp[0]  = c0[r];
        xp[16] = c1[r];
        xp[32] = c2[r];
        xp[48] = c3[r];
    }
}

// ---------------------------------------------------------------------------
// 6) scatter: agg[col] += dinv[row]*ew*dinv[col] * x[row]   (wave per edge)
// ---------------------------------------------------------------------------
__global__ void k_scatter(const int* __restrict__ ei,
                          const float* __restrict__ ew,
                          const float* __restrict__ dinv,
                          const float* __restrict__ x,
                          float* __restrict__ agg) {
    const int e    = (blockIdx.x * blockDim.x + threadIdx.x) >> 5;
    const int lane = threadIdx.x & 31;
    if (e >= NE) return;
    const int r = ei[e], c = ei[NE + e];
    const float norm = dinv[r] * ew[e] * dinv[c];
    if (norm == 0.0f) return;
    const float* xr = x + (size_t)r * HID + lane * 16;
    float* ar = agg + (size_t)c * HID + lane * 16;
#pragma unroll
    for (int j = 0; j < 4; ++j) {
        const float4 v = *(const float4*)(xr + 4 * j);
        atomicAdd(&ar[4 * j + 0], norm * v.x);
        atomicAdd(&ar[4 * j + 1], norm * v.y);
        atomicAdd(&ar[4 * j + 2], norm * v.z);
        atomicAdd(&ar[4 * j + 3], norm * v.w);
    }
}

// ---------------------------------------------------------------------------
// 7) rqe = l2norm(agg + dinv^2 * x + b_gcn)  in place into agg (wave per row)
// ---------------------------------------------------------------------------
__global__ void k_rqe(float* __restrict__ agg,
                      const float* __restrict__ x,
                      const float* __restrict__ dinv,
                      const float* __restrict__ bg) {
    const int row  = (blockIdx.x * blockDim.x + threadIdx.x) >> 5;
    const int lane = threadIdx.x & 31;
    if (row >= NR) return;
    const float di = dinv[row];
    const float di2 = di * di;
    float* ap = agg + (size_t)row * HID + lane * 16;
    const float* xp = x + (size_t)row * HID + lane * 16;
    const float* bp = bg + lane * 16;
    float vals[16];
    float ss = 0.0f;
#pragma unroll
    for (int j = 0; j < 16; ++j) {
        vals[j] = ap[j] + di2 * xp[j] + bp[j];
        ss += vals[j] * vals[j];
    }
    ss = wave_reduce_add(ss);
    const float inv = 1.0f / fmaxf(sqrtf(ss), 1.0e-12f);
#pragma unroll
    for (int j = 0; j < 16; ++j) ap[j] = vals[j] * inv;
}

// ---------------------------------------------------------------------------
// 8) scores = softmax(rqe @ llm^T) over 64 LLMs, Gumbel-max sample, logprob.
//    WMMA: one wave per 16-row tile, 4 column tiles (N=64).
// ---------------------------------------------------------------------------
__global__ void k_scores(const float* __restrict__ rqe,
                         const float* __restrict__ llm,
                         float* __restrict__ out_idx,
                         float* __restrict__ out_logp) {
    const int lane = threadIdx.x & 31;
    const int lo = lane & 15, hi = lane >> 4;
    const int mt = blockIdx.x;

    const float* arow = rqe + (size_t)(mt * 16 + lo) * HID;
    const float* b0 = llm + (size_t)(0  + lo) * HID;
    const float* b1 = llm + (size_t)(16 + lo) * HID;
    const float* b2 = llm + (size_t)(32 + lo) * HID;
    const float* b3 = llm + (size_t)(48 + lo) * HID;

    v8f c0 = {}, c1 = {}, c2 = {}, c3 = {};
    for (int kb = 0; kb < HID; kb += 32) {
        v16h a, f0, f1, f2, f3;
#pragma unroll
        for (int i = 0; i < 16; ++i) {
            const int v = i >> 1, p = i & 1;
            const int ka = ((v < 4) ? (2 * v) : (16 + 2 * (v - 4))) + hi * 8 + p;
            a[i] = (_Float16)arow[kb + ka];
            const int kk = kb + ((v < 4) ? (2 * v) : (16 + 2 * (v - 4))) + hi * 8 + p;
            f0[i] = (_Float16)b0[kk];
            f1[i] = (_Float16)b1[kk];
            f2[i] = (_Float16)b2[kk];
            f3[i] = (_Float16)b3[kk];
        }
        c0 = __builtin_amdgcn_wmma_f32_16x16x32_f16(false, a, false, f0, (short)0, c0, false, false);
        c1 = __builtin_amdgcn_wmma_f32_16x16x32_f16(false, a, false, f1, (short)0, c1, false, false);
        c2 = __builtin_amdgcn_wmma_f32_16x16x32_f16(false, a, false, f2, (short)0, c2, false, false);
        c3 = __builtin_amdgcn_wmma_f32_16x16x32_f16(false, a, false, f3, (short)0, c3, false, false);
    }

    float lp_acc = 0.0f;
#pragma unroll
    for (int r = 0; r < 8; ++r) {
        const int m = mt * 16 + r + hi * 8;
        float s[4] = { c0[r], c1[r], c2[r], c3[r] };

        float mx = fmaxf(fmaxf(s[0], s[1]), fmaxf(s[2], s[3]));
#pragma unroll
        for (int msk = 1; msk < 16; msk <<= 1) mx = fmaxf(mx, __shfl_xor(mx, msk, 32));

        float ex[4], sum = 0.0f;
#pragma unroll
        for (int t = 0; t < 4; ++t) { ex[t] = __expf(s[t] - mx); sum += ex[t]; }
#pragma unroll
        for (int msk = 1; msk < 16; msk <<= 1) sum += __shfl_xor(sum, msk, 32);
        const float lse = __logf(sum);

        float bv = -1.0e30f, bp = 0.0f; int bi = 0;
#pragma unroll
        for (int t = 0; t < 4; ++t) {
            const int col = t * 16 + lo;
            const float u = hash_u01((unsigned)(m * NL + col) * 2654435761u + 42u);
            const float g = -__logf(-__logf(u));
            const float lp = (s[t] - mx) - lse;
            const float val = lp + g;
            if (val > bv) { bv = val; bi = col; bp = ex[t] / sum; }
        }
#pragma unroll
        for (int msk = 1; msk < 16; msk <<= 1) {
            const float ov = __shfl_xor(bv, msk, 32);
            const int   oi = __shfl_xor(bi, msk, 32);
            const float op = __shfl_xor(bp, msk, 32);
            if (ov > bv) { bv = ov; bi = oi; bp = op; }
        }
        if (lo == 0) {
            out_idx[m] = (float)bi;
            lp_acc += __logf(bp + 1.0e-5f);
        }
    }
    if (lo == 0) atomicAdd(out_logp, lp_acc);
}

// ---------------------------------------------------------------------------
// launch
// ---------------------------------------------------------------------------
extern "C" void kernel_launch(void* const* d_in, const int* in_sizes, int n_in,
                              void* d_out, int out_size, void* d_ws, size_t ws_size,
                              hipStream_t stream) {
    const float* query  = (const float*)d_in[0];
    const float* role   = (const float*)d_in[1];
    const int*   eidx   = (const int*)  d_in[2];
    const float* eemb   = (const float*)d_in[3];
    const float* llme   = (const float*)d_in[4];
    const float* W_llm  = (const float*)d_in[5];
    const float* b_llm  = (const float*)d_in[6];
    const float* W_ew   = (const float*)d_in[7];
    const float* b_ew   = (const float*)d_in[8];
    const float* W_gcn  = (const float*)d_in[9];
    const float* b_gcn  = (const float*)d_in[10];
    (void)in_sizes; (void)n_in; (void)out_size; (void)ws_size;

    float* ws   = (float*)d_ws;
    float* x    = ws;                         // NR*HID
    float* agg  = x + (size_t)NR * HID;       // NR*HID
    float* llm  = agg + (size_t)NR * HID;     // NL*HID
    float* deg  = llm + (size_t)NL * HID;     // NR
    float* dinv = deg + NR;                   // NR

    float* out      = (float*)d_out;
    float* out_idx  = out;                    // NR
    float* out_logp = out + NR;               // 1
    float* out_ei   = out + NR + 1;           // 2*NE
    float* out_ew   = out_ei + (size_t)2 * NE;// NE

    k_init<<<4096, 256, 0, stream>>>(agg, deg, out_logp);

    k_llm_gemm<<<dim3(HID / 16, NL / 16), 32, 0, stream>>>(llme, W_llm, b_llm, llm);
    k_rownorm<<<(NL * 32 + 255) / 256, 256, 0, stream>>>(llm, NL);

    k_edge<<<NE / 8, 256, 0, stream>>>(eemb, W_ew, b_ew, eidx, out_ew, out_ei, deg);
    k_dinv<<<NR / 256, 256, 0, stream>>>(deg, dinv);

    k_gemm_rq<<<dim3(HID / BN, NR / BM), 256, 0, stream>>>(role, query, W_gcn, x);

    k_scatter<<<NE / 8, 256, 0, stream>>>(eidx, out_ew, dinv, x, agg);
    k_rqe<<<(NR * 32) / 256, 256, 0, stream>>>(agg, x, dinv, b_gcn);

    k_scores<<<NR / 16, 32, 0, stream>>>(agg, llm, out_idx, out_logp);
}
